// SimpleMultiHeadAttention_42966852829766
// MI455X (gfx1250) — matrix operand order repaired and checked by reading, not compiled, not run on hardware
//
#include <hip/hip_runtime.h>
#include <hip/hip_bf16.h>

// ---------------------------------------------------------------------------
// MHA forward for MI455X (gfx1250, wave32).
// All six GEMMs use v_wmma_f32_16x16x32_bf16 (fp32 accumulate).
// bf16 tiles are staged global->LDS with global_load_async_to_lds_b128 into a
// 2-deep LDS pipeline (ASYNCcnt in-order completion + s_wait_asynccnt N), so
// the DMA for tile i+1 overlaps the WMMAs on tile i. fp32 tiles are staged
// with vectorized pack + ds_store_b128 into the same double buffer.
// V is produced transposed (Vt[d][s]) so every bf16 operand is k-contiguous.
// ---------------------------------------------------------------------------

#define D_MODEL 1024
#define NUM_HEADS 16
#define DEPTH 64
#define BATCH 4
#define SEQ 2048

typedef __attribute__((ext_vector_type(16))) __bf16 v16bf;
typedef __attribute__((ext_vector_type(8)))  float  v8f;
typedef __attribute__((ext_vector_type(4)))  float  v4f;
typedef __attribute__((ext_vector_type(4)))  unsigned int v4u;

union Frag {
    v16bf v;
    v4u   u[2];
};

__device__ __forceinline__ unsigned short f32_to_bf16(float f) {
    union { float f; unsigned u; } x;
    x.f = f;
    unsigned u = x.u;
    u += 0x7FFFu + ((u >> 16) & 1u);   // round-to-nearest-even
    return (unsigned short)(u >> 16);
}
__device__ __forceinline__ unsigned pack_bf16x2(float lo, float hi) {
    return (unsigned)f32_to_bf16(lo) | ((unsigned)f32_to_bf16(hi) << 16);
}

// Async DMA: 16 bytes per lane, global -> LDS, no VGPR round trip.
// LDS operand = low 32 bits of the generic pointer (= LDS byte offset).
__device__ __forceinline__ void async_ld_b128(void* lds, const void* gp) {
    unsigned           lo = (unsigned)(unsigned long long)lds;
    unsigned long long ga = (unsigned long long)gp;
    asm volatile("global_load_async_to_lds_b128 %0, %1, off"
                 :: "v"(lo), "v"(ga) : "memory");
}

// ---------------------------------------------------------------------------
// Batched GEMM:  C = scale * (A @ B^T) + bias
//   A[m,k] at A + m*lda + k        (k contiguous; fp32 or bf16)
//   B[n,k] at B + n*sbn + k        (k contiguous; fp32 or bf16)
//   C_MODE: 0 = fp32 C[m*ldc+n], 1 = bf16 C[m*ldc+n], 2 = bf16 C[n*ldc+m]
// Block: 256 threads (8 waves). Tile 128(M) x 64(N); K pipelined 32 at a time
// through a 2-deep LDS double buffer. Wave (wm -> 32 rows, wn -> 32 cols):
// 4 accumulators, 4 WMMAs per K-step with 2 A-frags x 2 B-frags reuse.
// ---------------------------------------------------------------------------
template <bool A_BF16, bool B_BF16, int C_MODE>
__global__ __launch_bounds__(256) void gemm_wmma(
    const void* __restrict__ Av, long long aSB, long long aSH, int lda,
    const void* __restrict__ Bv, long long bSB, long long bSH, int sbn,
    void*       __restrict__ Cv, long long cSB, long long cSH, int ldc,
    const float* __restrict__ bias, float scale, int K, int H)
{
    __shared__ __align__(16) unsigned short Alds[2][128 * 32];   // 16 KB
    __shared__ __align__(16) unsigned short Blds[2][64 * 32];    //  8 KB

    constexpr int PER_ITER = (A_BF16 ? 2 : 0) + (B_BF16 ? 1 : 0);

    const int bz = blockIdx.z / H;
    const int hz = blockIdx.z % H;
    const long long aOff = (long long)bz * aSB + (long long)hz * aSH;
    const long long bOff = (long long)bz * bSB + (long long)hz * bSH;
    const long long cOff = (long long)bz * cSB + (long long)hz * cSH;

    const float*          Af = (const float*)Av + aOff;
    const unsigned short* Ah = (const unsigned short*)Av + aOff;
    const float*          Bf = (const float*)Bv + bOff;
    const unsigned short* Bh = (const unsigned short*)Bv + bOff;

    const int tid  = threadIdx.x;
    const int lane = tid & 31;
    const int wave = tid >> 5;
    const int wm   = wave & 3;
    const int wn   = wave >> 2;
    const int half = lane >> 4;
    const int l15  = lane & 15;

    const int m0 = blockIdx.y * 128;
    const int n0 = blockIdx.x * 64;

    // ---- tile stage: global -> LDS buf (async for bf16, pack for fp32) ---
    auto stage = [&](int buf, int t) {
        const int kk0 = t * 32;
        #pragma unroll
        for (int it = 0; it < 2; ++it) {
            const int idx = it * 2048 + tid * 8;
            const int row = idx >> 5;
            const int col = idx & 31;
            if constexpr (A_BF16) {
                async_ld_b128(&Alds[buf][row * 32 + col],
                              Ah + (size_t)(m0 + row) * lda + (kk0 + col));
            } else {
                const float* s = Af + (size_t)(m0 + row) * lda + (kk0 + col);
                v4f f0 = *(const v4f*)(s);
                v4f f1 = *(const v4f*)(s + 4);
                v4u d;
                d.x = pack_bf16x2(f0.x, f0.y);
                d.y = pack_bf16x2(f0.z, f0.w);
                d.z = pack_bf16x2(f1.x, f1.y);
                d.w = pack_bf16x2(f1.z, f1.w);
                *(v4u*)&Alds[buf][row * 32 + col] = d;
            }
        }
        {
            const int idx = tid * 8;
            const int n   = idx >> 5;
            const int col = idx & 31;
            if constexpr (B_BF16) {
                async_ld_b128(&Blds[buf][n * 32 + col],
                              Bh + (size_t)(n0 + n) * sbn + (kk0 + col));
            } else {
                const float* s = Bf + (size_t)(n0 + n) * sbn + (kk0 + col);
                v4f f0 = *(const v4f*)(s);
                v4f f1 = *(const v4f*)(s + 4);
                v4u d;
                d.x = pack_bf16x2(f0.x, f0.y);
                d.y = pack_bf16x2(f0.z, f0.w);
                d.z = pack_bf16x2(f1.x, f1.y);
                d.w = pack_bf16x2(f1.z, f1.w);
                *(v4u*)&Blds[buf][n * 32 + col] = d;
            }
        }
    };

    const int NT = K >> 5;          // K tiles (all call sites: NT >= 2)
    v8f acc[2][2] = {};

    stage(0, 0);
    if (NT > 1) stage(1, 1);

    for (int i = 0; i < NT; ++i) {
        // Tile i is resident when at most tile i+1's async issues remain
        // outstanding (ASYNCcnt completes in order).
        if constexpr (PER_ITER > 0) {
            if (i + 1 < NT)
                asm volatile("s_wait_asynccnt %0" :: "n"(PER_ITER) : "memory");
            else
                asm volatile("s_wait_asynccnt 0" ::: "memory");
        }
        __syncthreads();

        const int buf = i & 1;

        // ---- fragments per CDNA5 16-bit WMMA VGPR layout -----------------
        Frag a[2], b[2];
        #pragma unroll
        for (int mi = 0; mi < 2; ++mi) {
            const unsigned short* Ar = &Alds[buf][(wm * 32 + mi * 16 + l15) * 32];
            a[mi].u[0] = *(const v4u*)(Ar + half * 8);
            a[mi].u[1] = *(const v4u*)(Ar + 16 + half * 8);
        }
        #pragma unroll
        for (int ni = 0; ni < 2; ++ni) {
            const unsigned short* Br =
                &Blds[buf][(wn * 32 + ni * 16 + l15) * 32];
            b[ni].u[0] = *(const v4u*)(Br + half * 8);
            b[ni].u[1] = *(const v4u*)(Br + 16 + half * 8);
        }
        #pragma unroll
        for (int mi = 0; mi < 2; ++mi)
            #pragma unroll
            for (int ni = 0; ni < 2; ++ni)
                acc[mi][ni] = __builtin_amdgcn_wmma_f32_16x16x32_bf16(
                    false, a[mi].v, false, b[ni].v, (short)0, acc[mi][ni],
                    false, false);

        __syncthreads();   // all waves done reading buf before refilling it

        if (i + 2 < NT) stage(buf, i + 2);
    }

    // ---- epilogue --------------------------------------------------------
    if constexpr (C_MODE == 2) {
        // bf16, transposed: C[n*ldc + m]; 8 acc rows = 8 contiguous m -> b128
        unsigned short* Ch = (unsigned short*)Cv + cOff;
        #pragma unroll
        for (int mi = 0; mi < 2; ++mi) {
            const int m_base = m0 + wm * 32 + mi * 16 + half * 8;
            #pragma unroll
            for (int ni = 0; ni < 2; ++ni) {
                const int n = n0 + wn * 32 + ni * 16 + l15;
                const float bb = bias ? bias[n] : 0.0f;
                const v8f va = acc[mi][ni];
                v4u d;
                d.x = pack_bf16x2(va[0] * scale + bb, va[1] * scale + bb);
                d.y = pack_bf16x2(va[2] * scale + bb, va[3] * scale + bb);
                d.z = pack_bf16x2(va[4] * scale + bb, va[5] * scale + bb);
                d.w = pack_bf16x2(va[6] * scale + bb, va[7] * scale + bb);
                *(v4u*)(Ch + (size_t)n * ldc + m_base) = d;
            }
        }
    } else {
        float*          Cf = (float*)Cv + cOff;
        unsigned short* Ch = (unsigned short*)Cv + cOff;
        #pragma unroll
        for (int mi = 0; mi < 2; ++mi) {
            const int m_base = m0 + wm * 32 + mi * 16 + half * 8;
            #pragma unroll
            for (int ni = 0; ni < 2; ++ni) {
                const int n = n0 + wn * 32 + ni * 16 + l15;
                const float bb = bias ? bias[n] : 0.0f;
                #pragma unroll
                for (int r = 0; r < 8; ++r) {
                    const float val = acc[mi][ni][r] * scale + bb;
                    const size_t off = (size_t)(m_base + r) * ldc + n;
                    if constexpr (C_MODE == 1) Ch[off] = f32_to_bf16(val);
                    else                       Cf[off] = val;
                }
            }
        }
    }
}

// ---------------------------------------------------------------------------
// Row softmax, in place. One 256-thread block per row of length n (=2048).
// ---------------------------------------------------------------------------
__global__ __launch_bounds__(256) void softmax_rows(float* __restrict__ attn, int n)
{
    __shared__ float red[256];
    float* p = attn + (size_t)blockIdx.x * n;
    const int tid = threadIdx.x;

    float m = -INFINITY;
    for (int i = tid; i < n; i += 256) m = fmaxf(m, p[i]);
    red[tid] = m;
    __syncthreads();
    for (int s = 128; s > 0; s >>= 1) {
        if (tid < s) red[tid] = fmaxf(red[tid], red[tid + s]);
        __syncthreads();
    }
    const float rowmax = red[0];
    __syncthreads();

    float sum = 0.f;
    for (int i = tid; i < n; i += 256) {
        float e = __expf(p[i] - rowmax);
        p[i] = e;
        sum += e;
    }
    red[tid] = sum;
    __syncthreads();
    for (int s = 128; s > 0; s >>= 1) {
        if (tid < s) red[tid] += red[tid + s];
        __syncthreads();
    }
    const float inv = 1.0f / red[0];
    __syncthreads();
    for (int i = tid; i < n; i += 256) p[i] *= inv;
}

// ---------------------------------------------------------------------------
extern "C" void kernel_launch(void* const* d_in, const int* in_sizes, int n_in,
                              void* d_out, int out_size, void* d_ws, size_t ws_size,
                              hipStream_t stream)
{
    (void)in_sizes; (void)n_in; (void)out_size; (void)ws_size;

    const float* q  = (const float*)d_in[0];
    const float* k  = (const float*)d_in[1];
    const float* v  = (const float*)d_in[2];
    const float* wq = (const float*)d_in[3];
    const float* bq = (const float*)d_in[4];
    const float* wk = (const float*)d_in[5];
    const float* bk = (const float*)d_in[6];
    const float* wv = (const float*)d_in[7];
    const float* bv = (const float*)d_in[8];
    const float* wo = (const float*)d_in[9];
    const float* bo = (const float*)d_in[10];

    const long long BSD = (long long)BATCH * SEQ * D_MODEL;   // 8,388,608
    const long long MS  = (long long)BATCH * SEQ;             // 8192
    // bf16 intermediates in workspace
    unsigned short* Qp  = (unsigned short*)d_ws;              // [B*S][1024]
    unsigned short* Kp  = Qp + BSD;                           // [B*S][1024]
    unsigned short* Vt  = Kp + BSD;                           // [1024][B*S] (transposed)
    unsigned short* Ctx = Vt + BSD;                           // [B*S][1024]

    float* out  = (float*)d_out;        // [B,S,D]
    float* attn = out + BSD;            // [B,H,S,S]

    const dim3 blk(256);

    // --- Q/K projections: 8192 x 1024 x 1024, fp32 in -> bf16 out ---------
    const dim3 gProj(D_MODEL / 64, MS / 128, 1);
    gemm_wmma<false, false, 1><<<gProj, blk, 0, stream>>>(
        q, 0, 0, D_MODEL, wq, 0, 0, D_MODEL,
        Qp, 0, 0, D_MODEL, bq, 1.0f, D_MODEL, 1);
    gemm_wmma<false, false, 1><<<gProj, blk, 0, stream>>>(
        k, 0, 0, D_MODEL, wk, 0, 0, D_MODEL,
        Kp, 0, 0, D_MODEL, bk, 1.0f, D_MODEL, 1);
    // --- V projection, written transposed: Vt[d_total][B*S] bf16 ----------
    gemm_wmma<false, false, 2><<<gProj, blk, 0, stream>>>(
        v, 0, 0, D_MODEL, wv, 0, 0, D_MODEL,
        Vt, 0, 0, (int)MS, bv, 1.0f, D_MODEL, 1);

    // --- scores = Qh @ Kh^T / 8: per (b,h), 2048 x 2048 x 64 (async bf16) -
    const dim3 gScore(SEQ / 64, SEQ / 128, BATCH * NUM_HEADS);
    gemm_wmma<true, true, 0><<<gScore, blk, 0, stream>>>(
        Qp, (long long)SEQ * D_MODEL, DEPTH, D_MODEL,
        Kp, (long long)SEQ * D_MODEL, DEPTH, D_MODEL,
        attn, (long long)NUM_HEADS * SEQ * SEQ, (long long)SEQ * SEQ, SEQ,
        nullptr, 0.125f, DEPTH, NUM_HEADS);

    // --- softmax over rows of attn ----------------------------------------
    softmax_rows<<<dim3(BATCH * NUM_HEADS * SEQ), blk, 0, stream>>>(attn, SEQ);

    // --- ctx = attn @ Vh: per (b,h), 2048 x 64 x 2048 ---------------------
    //     B = Vt rows (k-contiguous): element (d,s) at Vt[(h*64+d)*MS + b*S + s]
    const dim3 gAV(DEPTH / 64, SEQ / 128, BATCH * NUM_HEADS);
    gemm_wmma<false, true, 1><<<gAV, blk, 0, stream>>>(
        attn, (long long)NUM_HEADS * SEQ * SEQ, (long long)SEQ * SEQ, SEQ,
        Vt, (long long)SEQ, (long long)DEPTH * MS, (int)MS,
        Ctx, (long long)SEQ * D_MODEL, DEPTH, D_MODEL,
        nullptr, 1.0f, SEQ, NUM_HEADS);

    // --- output projection: out = Ctx @ Wo^T + bo (bf16 A async) ----------
    gemm_wmma<true, false, 0><<<gProj, blk, 0, stream>>>(
        Ctx, 0, 0, D_MODEL, wo, 0, 0, D_MODEL,
        out, 0, 0, D_MODEL, bo, 1.0f, D_MODEL, 1);
}
